// CSWin3DBlock_63814624084284
// MI455X (gfx1250) — matrix operand order repaired and checked by reading, not compiled, not run on hardware
//
#include <hip/hip_runtime.h>
#include <hip/hip_bf16.h>

typedef __attribute__((ext_vector_type(16))) _Float16 v16h;
typedef __attribute__((ext_vector_type(8)))  float    v8f;
typedef __attribute__((ext_vector_type(4)))  unsigned int u32x4;
typedef __attribute__((ext_vector_type(8)))  int      i32x8;
typedef __attribute__((ext_vector_type(4)))  int      i32x4;

union AFrag { v16h v; _Float16 e[16]; };

// Geometry constants
#define BT   4      // batch
#define TT   8      // temporal
#define RES  32
#define CC   128
#define LTOT 32768  // B*T*RES*RES
#define HIDN 512

__device__ __forceinline__ int token_index(int b, int wt, int wh, int ww, int p,
                                           int Hsp, int Wsp) {
  int hw = Hsp * Wsp;
  int t_in = p / hw;
  int rem  = p - t_in * hw;
  int h_in = rem / Wsp;
  int w_in = rem - h_in * Wsp;
  int t = wt * 4 + t_in;
  int h = wh * Hsp + h_in;
  int w = ww * Wsp + w_in;
  return ((b * TT + t) * RES + h) * RES + w;
}

// ---------------------------------------------------------------------------
// Tensor Data Mover: 4D tile (f16 elements) from global -> LDS.
// tile0 = contiguous elements; tile1/2/3 outer dims with strides s0/s1/s2
// (in elements). Pass tile1..3 = 0 for lower-dim tiles.
// D# bitfields per CDNA5 ISA 8.3-8.6 (count=1, type=2=image, data_size=2B).
// Toolchain uses the 6-arg builtin: (g0, g1, g2, g3, extra, cpol).
// ---------------------------------------------------------------------------
__device__ __forceinline__ void tdm_load_f16_4d(
    unsigned lds_off, const void* gptr,
    unsigned tile0, unsigned tile1, unsigned tile2, unsigned tile3,
    unsigned long long s0, unsigned long long s1, unsigned long long s2) {
  unsigned long long ga = (unsigned long long)gptr;
  u32x4 g0;
  g0[0] = 1u;                                   // count=1, user descriptor
  g0[1] = lds_off;                              // lds_addr (bytes)
  g0[2] = (unsigned)(ga & 0xffffffffu);         // global_addr[31:0]
  g0[3] = (unsigned)((ga >> 32) & 0x1ffffffu)   // global_addr[56:32]
          | (2u << 30);                         // type = 2 (image)
  i32x8 g1;
  g1[0] = (int)(1u << 16);                      // data_size = 1 (2 bytes)
  g1[1] = (int)((tile0 & 0xffffu) << 16);       // tensor_dim0[15:0] @ [63:48]
  g1[2] = (int)(((tile0 >> 16) & 0xffffu)       // tensor_dim0[31:16]
          | ((tile1 & 0xffffu) << 16));         // tensor_dim1[15:0]
  g1[3] = (int)(((tile1 >> 16) & 0xffffu)       // tensor_dim1[31:16]
          | ((tile0 & 0xffffu) << 16));         // tile_dim0
  g1[4] = (int)((tile1 & 0xffffu)               // tile_dim1
          | ((tile2 & 0xffffu) << 16));         // tile_dim2
  g1[5] = (int)(unsigned)(s0 & 0xffffffffu);    // tensor_dim0_stride[31:0]
  g1[6] = (int)((unsigned)((s0 >> 32) & 0xffffu)   // dim0_stride[47:32]
          | ((unsigned)(s1 & 0xffffu) << 16));     // dim1_stride[15:0]
  g1[7] = (int)(unsigned)((s1 >> 16) & 0xffffffffu); // dim1_stride[47:16]
  i32x4 g2;
  g2[0] = (int)tile2;                           // tensor_dim2
  g2[1] = (int)tile3;                           // tensor_dim3
  g2[2] = (int)(unsigned)(s2 & 0xffffffffu);    // tensor_dim2_stride[31:0]
  g2[3] = (int)((unsigned)((s2 >> 32) & 0xffffu)
          | ((tile3 & 0xffffu) << 16));         // tile_dim3
  i32x4 g3;
  g3[0] = 0; g3[1] = 0; g3[2] = 0; g3[3] = 0;   // no dim3 stride / dim4
  i32x8 g4 = {0, 0, 0, 0, 0, 0, 0, 0};          // unused extra operand
  __builtin_amdgcn_tensor_load_to_lds(g0, g1, g2, g3, g4, 0);
}

__device__ __forceinline__ unsigned lds_offset_of(const void* p) {
  // Generic LDS addresses carry the LDS byte offset in the low 32 bits.
  return (unsigned)(unsigned long long)p;
}

// ---------------------------------------------------------------------------
// Weight conversion: f32 -> f16; fold attention scale (0.25) into q rows.
// ---------------------------------------------------------------------------
__global__ __launch_bounds__(256) void wconv_kernel(
    const float* __restrict__ qkv_w, const float* __restrict__ proj_w,
    const float* __restrict__ fc1_w, const float* __restrict__ fc2_w,
    _Float16* __restrict__ Wq, _Float16* __restrict__ Wp,
    _Float16* __restrict__ W1, _Float16* __restrict__ W2) {
  int i = blockIdx.x * 256 + threadIdx.x;
  if (i < 384 * 128) {
    float v = qkv_w[i];
    if (i < 128 * 128) v *= 0.25f;  // q scale = hd^-0.5 = 0.25
    Wq[i] = (_Float16)v;
  }
  if (i < 128 * 128) Wp[i] = (_Float16)proj_w[i];
  if (i < 512 * 128) W1[i] = (_Float16)fc1_w[i];
  if (i < 128 * 512) W2[i] = (_Float16)fc2_w[i];
}

// ---------------------------------------------------------------------------
// Fused LayerNorm1 + QKV GEMM (M=32768, N=384, K=128), f16 WMMA.
// ---------------------------------------------------------------------------
__global__ __launch_bounds__(256) void ln_qkv_kernel(
    const float* __restrict__ x, const float* __restrict__ lw,
    const float* __restrict__ lb, const _Float16* __restrict__ Wq,
    _Float16* __restrict__ qkv) {
  __shared__ _Float16 Xs[16 * 128];
  const int tid = threadIdx.x;
  const int row = tid >> 4;
  const int l16 = tid & 15;
  const size_t grow = (size_t)blockIdx.x * 16 + row;

  float vals[8];
  float sum = 0.f, sq = 0.f;
#pragma unroll
  for (int j = 0; j < 8; ++j) {
    float v = x[grow * CC + l16 * 8 + j];
    vals[j] = v; sum += v; sq += v * v;
  }
#pragma unroll
  for (int m = 1; m < 16; m <<= 1) {
    sum += __shfl_xor(sum, m, 32);
    sq  += __shfl_xor(sq,  m, 32);
  }
  float mu   = sum * (1.f / 128.f);
  float var  = sq * (1.f / 128.f) - mu * mu;
  float rstd = rsqrtf(var + 1e-5f);
#pragma unroll
  for (int j = 0; j < 8; ++j) {
    int c = l16 * 8 + j;
    Xs[row * 128 + c] = (_Float16)((vals[j] - mu) * rstd * lw[c] + lb[c]);
  }
  __syncthreads();

  const int wave = tid >> 5, lane = tid & 31;
  const int r16 = lane & 15, half = lane >> 4;
#pragma unroll
  for (int w3 = 0; w3 < 3; ++w3) {
    int nt = wave * 3 + w3;
    v8f acc = {};
#pragma unroll
    for (int kt = 0; kt < 4; ++kt) {
      AFrag a, bf;
      const _Float16* ap = &Xs[r16 * 128 + kt * 32 + half * 8];
#pragma unroll
      for (int j = 0; j < 8; ++j) { a.e[j] = ap[j]; a.e[8 + j] = ap[16 + j]; }
      const _Float16* bp = Wq + (size_t)(nt * 16 + r16) * 128 + kt * 32 + half * 8;
#pragma unroll
      for (int j = 0; j < 8; ++j) { bf.e[j] = bp[j]; bf.e[8 + j] = bp[16 + j]; }
      acc = __builtin_amdgcn_wmma_f32_16x16x32_f16(false, a.v, false, bf.v,
                                                   (short)0, acc, false, false);
    }
    int col = nt * 16 + r16;
#pragma unroll
    for (int i = 0; i < 8; ++i) {
      size_t orow = (size_t)blockIdx.x * 16 + half * 8 + i;
      qkv[orow * 384 + col] = (_Float16)acc[i];
    }
  }
}

// ---------------------------------------------------------------------------
// Windowed flash attention: one block per (window, head). win=512, hd=16.
// K tile is fetched by the Tensor Data Mover as a 4D tile while the waves
// build the transposed V tile; then flash WMMA loop.
// ---------------------------------------------------------------------------
__global__ __launch_bounds__(256) void attn_kernel(
    const _Float16* __restrict__ qkv, float* __restrict__ att,
    int Hsp, int Wsp, int chan_off) {
  __shared__ _Float16 Ks[512 * 16];     // [key][hd]   (TDM-filled)
  __shared__ _Float16 Vt[16 * 512];     // [hd][key]   (transposed)
  __shared__ _Float16 Ps[8][16 * 32];   // per-wave P scratch

  const int head = blockIdx.y;
  const int wid  = blockIdx.x;
  const int nWH = RES / Hsp, nWW = RES / Wsp;
  const int wpb = 2 * nWH * nWW;        // windows per batch (=16)
  const int b = wid / wpb;
  int rem = wid - b * wpb;
  const int wt = rem / (nWH * nWW);
  rem -= wt * (nWH * nWW);
  const int wh = rem / nWW;
  const int ww = rem - wh * nWW;
  const int ch = chan_off + head * 16;
  const int tid = threadIdx.x;

  // TDM: K window tile (16ch x Wsp x Hsp x 4t) -> Ks, issued by wave 0.
  {
    int l0 = token_index(b, wt, wh, ww, 0, Hsp, Wsp);
    if (tid < 32) {
      tdm_load_f16_4d(lds_offset_of(Ks), qkv + (size_t)l0 * 384 + 128 + ch,
                      16u, (unsigned)Wsp, (unsigned)Hsp, 4u,
                      384ull, 384ull * RES, 384ull * RES * RES);
    }
  }
  // Meanwhile: transposed V staging (TDM cannot transpose).
  for (int p = tid; p < 512; p += 256) {
    int l = token_index(b, wt, wh, ww, p, Hsp, Wsp);
    const _Float16* vrow = qkv + (size_t)l * 384 + 256 + ch;
#pragma unroll
    for (int j = 0; j < 16; ++j) Vt[j * 512 + p] = vrow[j];
  }
  if (tid < 32) __builtin_amdgcn_s_wait_tensorcnt(0);
  __syncthreads();

  const int wave = tid >> 5, lane = tid & 31;
  const int r16 = lane & 15, half = lane >> 4;

  for (int qt = wave; qt < 32; qt += 8) {
    AFrag qa;
    {
      int p = qt * 16 + r16;
      int l = token_index(b, wt, wh, ww, p, Hsp, Wsp);
      const _Float16* qrow = qkv + (size_t)l * 384 + ch;
#pragma unroll
      for (int j = 0; j < 8; ++j) { qa.e[j] = qrow[j]; qa.e[8 + j] = qrow[8 + j]; }
    }
    float m[8], s[8];
    v8f O = {};
#pragma unroll
    for (int i = 0; i < 8; ++i) { m[i] = -1e30f; s[i] = 0.f; }

    for (int kb = 0; kb < 16; ++kb) {
      AFrag kb0, kb1;
#pragma unroll
      for (int j = 0; j < 16; ++j) { kb0.e[j] = (_Float16)0.f; kb1.e[j] = (_Float16)0.f; }
      if (half == 0) {
        const _Float16* k0 = &Ks[(kb * 32 + r16) * 16];
        const _Float16* k1 = &Ks[(kb * 32 + 16 + r16) * 16];
#pragma unroll
        for (int j = 0; j < 16; ++j) { kb0.e[j] = k0[j]; kb1.e[j] = k1[j]; }
      }
      v8f z = {};
      v8f S0 = __builtin_amdgcn_wmma_f32_16x16x32_f16(false, qa.v, false, kb0.v,
                                                      (short)0, z, false, false);
      v8f S1 = __builtin_amdgcn_wmma_f32_16x16x32_f16(false, qa.v, false, kb1.v,
                                                      (short)0, z, false, false);
#pragma unroll
      for (int i = 0; i < 8; ++i) {
        float t = fmaxf(S0[i], S1[i]);
#pragma unroll
        for (int msk = 1; msk < 16; msk <<= 1) t = fmaxf(t, __shfl_xor(t, msk, 32));
        float mn = fmaxf(m[i], t);
        float alpha = __expf(m[i] - mn);
        float p0 = __expf(S0[i] - mn);
        float p1 = __expf(S1[i] - mn);
        float r = p0 + p1;
#pragma unroll
        for (int msk = 1; msk < 16; msk <<= 1) r += __shfl_xor(r, msk, 32);
        s[i] = s[i] * alpha + r;
        m[i] = mn;
        O[i] *= alpha;
        int prow = half * 8 + i;
        Ps[wave][prow * 32 + r16]      = (_Float16)p0;
        Ps[wave][prow * 32 + 16 + r16] = (_Float16)p1;
      }
      AFrag pa;
      {
        const _Float16* pr = &Ps[wave][r16 * 32];
#pragma unroll
        for (int j = 0; j < 8; ++j) {
          pa.e[j]     = pr[half * 8 + j];
          pa.e[8 + j] = pr[16 + half * 8 + j];
        }
      }
      AFrag vb;
      {
        const _Float16* vp = &Vt[r16 * 512 + kb * 32 + half * 8];
#pragma unroll
        for (int j = 0; j < 8; ++j) { vb.e[j] = vp[j]; vb.e[8 + j] = vp[16 + j]; }
      }
      O = __builtin_amdgcn_wmma_f32_16x16x32_f16(false, pa.v, false, vb.v,
                                                 (short)0, O, false, false);
    }
#pragma unroll
    for (int i = 0; i < 8; ++i) {
      int prow = half * 8 + i;
      int p = qt * 16 + prow;
      int l = token_index(b, wt, wh, ww, p, Hsp, Wsp);
      att[(size_t)l * CC + ch + r16] = O[i] / s[i];
    }
  }
}

// ---------------------------------------------------------------------------
// LePE: depthwise 3x3x3 conv over each window's V, added into att.
// V window staged by TDM in position-major layout Vl[p][c] (32 ch / chunk).
// ---------------------------------------------------------------------------
__global__ __launch_bounds__(256) void lepe_kernel(
    const _Float16* __restrict__ qkv, const float* __restrict__ cw,
    const float* __restrict__ cb, float* __restrict__ att,
    int Hsp, int Wsp, int chan_off) {
  __shared__ _Float16 Vl[512 * 32];   // [pos][chan] : matches TDM write order
  const int wid = blockIdx.x;
  const int nWH = RES / Hsp, nWW = RES / Wsp;
  const int wpb = 2 * nWH * nWW;
  const int b = wid / wpb;
  int rem = wid - b * wpb;
  const int wt = rem / (nWH * nWW);
  rem -= wt * (nWH * nWW);
  const int wh = rem / nWW;
  const int ww = rem - wh * nWW;
  const int tid = threadIdx.x;
  const int HW = Hsp * Wsp;
  const int l0 = token_index(b, wt, wh, ww, 0, Hsp, Wsp);

  for (int cc = 0; cc < 2; ++cc) {
    if (tid < 32) {
      tdm_load_f16_4d(lds_offset_of(Vl),
                      qkv + (size_t)l0 * 384 + 256 + chan_off + cc * 32,
                      32u, (unsigned)Wsp, (unsigned)Hsp, 4u,
                      384ull, 384ull * RES, 384ull * RES * RES);
      __builtin_amdgcn_s_wait_tensorcnt(0);
    }
    __syncthreads();
    for (int idx = tid; idx < 32 * 512; idx += 256) {
      int c = idx & 31, p = idx >> 5;
      int gc = cc * 32 + c;
      int t = p / HW, h = (p / Wsp) % Hsp, w0 = p % Wsp;
      float acc = cb[gc];
#pragma unroll
      for (int dt = -1; dt <= 1; ++dt)
#pragma unroll
        for (int dh = -1; dh <= 1; ++dh)
#pragma unroll
          for (int dw = -1; dw <= 1; ++dw) {
            int tt = t + dt, hh = h + dh, wb = w0 + dw;
            if ((unsigned)tt < 4u && (unsigned)hh < (unsigned)Hsp &&
                (unsigned)wb < (unsigned)Wsp) {
              float wv = cw[gc * 27 + (dt + 1) * 9 + (dh + 1) * 3 + (dw + 1)];
              acc += wv * (float)Vl[((tt * Hsp + hh) * Wsp + wb) * 32 + c];
            }
          }
      int l = token_index(b, wt, wh, ww, p, Hsp, Wsp);
      att[(size_t)l * CC + chan_off + gc] += acc;
    }
    __syncthreads();
  }
}

// ---------------------------------------------------------------------------
// Projection + residual: x2 = x + att @ Wp^T + pb  (writes d_out)
// ---------------------------------------------------------------------------
__global__ __launch_bounds__(256) void proj_kernel(
    const float* __restrict__ att, const _Float16* __restrict__ Wp,
    const float* __restrict__ pb, const float* __restrict__ x,
    float* __restrict__ x2) {
  __shared__ _Float16 As[16 * 128];
  const int tid = threadIdx.x;
  for (int idx = tid; idx < 2048; idx += 256)
    As[idx] = (_Float16)att[(size_t)blockIdx.x * 2048 + idx];
  __syncthreads();
  const int wave = tid >> 5, lane = tid & 31;
  const int r16 = lane & 15, half = lane >> 4;
  const int nt = wave;
  v8f acc = {};
#pragma unroll
  for (int kt = 0; kt < 4; ++kt) {
    AFrag a, bf;
    const _Float16* ap = &As[r16 * 128 + kt * 32 + half * 8];
#pragma unroll
    for (int j = 0; j < 8; ++j) { a.e[j] = ap[j]; a.e[8 + j] = ap[16 + j]; }
    const _Float16* bp = Wp + (size_t)(nt * 16 + r16) * 128 + kt * 32 + half * 8;
#pragma unroll
    for (int j = 0; j < 8; ++j) { bf.e[j] = bp[j]; bf.e[8 + j] = bp[16 + j]; }
    acc = __builtin_amdgcn_wmma_f32_16x16x32_f16(false, a.v, false, bf.v,
                                                 (short)0, acc, false, false);
  }
  int col = nt * 16 + r16;
#pragma unroll
  for (int i = 0; i < 8; ++i) {
    size_t orow = (size_t)blockIdx.x * 16 + half * 8 + i;
    x2[orow * CC + col] = x[orow * CC + col] + acc[i] + pb[col];
  }
}

// ---------------------------------------------------------------------------
// LN2 + FC1 + exact GELU (M=32768, N=512, K=128) -> h1 (f16)
// ---------------------------------------------------------------------------
__global__ __launch_bounds__(256) void ln2_fc1_kernel(
    const float* __restrict__ x2, const float* __restrict__ lw,
    const float* __restrict__ lb, const _Float16* __restrict__ W1,
    const float* __restrict__ b1, _Float16* __restrict__ h1) {
  __shared__ _Float16 Xs[16 * 128];
  const int tid = threadIdx.x;
  const int row = tid >> 4, l16 = tid & 15;
  const size_t grow = (size_t)blockIdx.x * 16 + row;
  float vals[8];
  float sum = 0.f, sq = 0.f;
#pragma unroll
  for (int j = 0; j < 8; ++j) {
    float v = x2[grow * CC + l16 * 8 + j];
    vals[j] = v; sum += v; sq += v * v;
  }
#pragma unroll
  for (int m = 1; m < 16; m <<= 1) {
    sum += __shfl_xor(sum, m, 32);
    sq  += __shfl_xor(sq,  m, 32);
  }
  float mu   = sum * (1.f / 128.f);
  float var  = sq * (1.f / 128.f) - mu * mu;
  float rstd = rsqrtf(var + 1e-5f);
#pragma unroll
  for (int j = 0; j < 8; ++j) {
    int c = l16 * 8 + j;
    Xs[row * 128 + c] = (_Float16)((vals[j] - mu) * rstd * lw[c] + lb[c]);
  }
  __syncthreads();

  const int wave = tid >> 5, lane = tid & 31;
  const int r16 = lane & 15, half = lane >> 4;
#pragma unroll
  for (int w4 = 0; w4 < 4; ++w4) {
    int nt = wave * 4 + w4;
    v8f acc = {};
#pragma unroll
    for (int kt = 0; kt < 4; ++kt) {
      AFrag a, bf;
      const _Float16* ap = &Xs[r16 * 128 + kt * 32 + half * 8];
#pragma unroll
      for (int j = 0; j < 8; ++j) { a.e[j] = ap[j]; a.e[8 + j] = ap[16 + j]; }
      const _Float16* bp = W1 + (size_t)(nt * 16 + r16) * 128 + kt * 32 + half * 8;
#pragma unroll
      for (int j = 0; j < 8; ++j) { bf.e[j] = bp[j]; bf.e[8 + j] = bp[16 + j]; }
      acc = __builtin_amdgcn_wmma_f32_16x16x32_f16(false, a.v, false, bf.v,
                                                   (short)0, acc, false, false);
    }
    int col = nt * 16 + r16;
#pragma unroll
    for (int i = 0; i < 8; ++i) {
      size_t orow = (size_t)blockIdx.x * 16 + half * 8 + i;
      float g = acc[i] + b1[col];
      float gl = 0.5f * g * (1.f + erff(g * 0.70710678118654752f));
      h1[orow * HIDN + col] = (_Float16)gl;
    }
  }
}

// ---------------------------------------------------------------------------
// FC2 + residual: out += h1 @ W2^T + b2   (out already holds x2)
// A-tile (16x512 f16, 16KB contiguous) staged by a single 1D TDM load.
// ---------------------------------------------------------------------------
__global__ __launch_bounds__(256) void fc2_kernel(
    const _Float16* __restrict__ h1, const _Float16* __restrict__ W2,
    const float* __restrict__ b2, float* __restrict__ out) {
  __shared__ _Float16 Hs[16 * 512];
  const int tid = threadIdx.x;
  if (tid < 32) {
    tdm_load_f16_4d(lds_offset_of(Hs), h1 + (size_t)blockIdx.x * 16 * 512,
                    8192u, 0u, 0u, 0u, 0ull, 0ull, 0ull);
    __builtin_amdgcn_s_wait_tensorcnt(0);
  }
  __syncthreads();
  const int wave = tid >> 5, lane = tid & 31;
  const int r16 = lane & 15, half = lane >> 4;
  const int nt = wave;
  v8f acc = {};
  for (int kt = 0; kt < 16; ++kt) {
    AFrag a, bf;
    const _Float16* ap = &Hs[r16 * 512 + kt * 32 + half * 8];
#pragma unroll
    for (int j = 0; j < 8; ++j) { a.e[j] = ap[j]; a.e[8 + j] = ap[16 + j]; }
    const _Float16* bp = W2 + (size_t)(nt * 16 + r16) * 512 + kt * 32 + half * 8;
#pragma unroll
    for (int j = 0; j < 8; ++j) { bf.e[j] = bp[j]; bf.e[8 + j] = bp[16 + j]; }
    acc = __builtin_amdgcn_wmma_f32_16x16x32_f16(false, a.v, false, bf.v,
                                                 (short)0, acc, false, false);
  }
  int col = nt * 16 + r16;
#pragma unroll
  for (int i = 0; i < 8; ++i) {
    size_t orow = (size_t)blockIdx.x * 16 + half * 8 + i;
    out[orow * CC + col] += acc[i] + b2[col];
  }
}

// ---------------------------------------------------------------------------
extern "C" void kernel_launch(void* const* d_in, const int* in_sizes, int n_in,
                              void* d_out, int out_size, void* d_ws, size_t ws_size,
                              hipStream_t stream) {
  const float* x      = (const float*)d_in[0];
  const float* n1w    = (const float*)d_in[1];
  const float* n1b    = (const float*)d_in[2];
  const float* qkv_w  = (const float*)d_in[3];
  const float* cw0    = (const float*)d_in[4];
  const float* cb0    = (const float*)d_in[5];
  const float* cw1    = (const float*)d_in[6];
  const float* cb1    = (const float*)d_in[7];
  const float* proj_w = (const float*)d_in[8];
  const float* proj_b = (const float*)d_in[9];
  const float* n2w    = (const float*)d_in[10];
  const float* n2b    = (const float*)d_in[11];
  const float* fc1_w  = (const float*)d_in[12];
  const float* fc1_b  = (const float*)d_in[13];
  const float* fc2_w  = (const float*)d_in[14];
  const float* fc2_b  = (const float*)d_in[15];
  float* out = (float*)d_out;

  char* ws = (char*)d_ws;
  size_t off = 0;
  auto alloc = [&](size_t bytes) -> void* {
    void* p = ws + off;
    off += (bytes + 255) & ~(size_t)255;
    return p;
  };
  _Float16* Wq  = (_Float16*)alloc((size_t)384 * 128 * 2);
  _Float16* Wp  = (_Float16*)alloc((size_t)128 * 128 * 2);
  _Float16* W1  = (_Float16*)alloc((size_t)512 * 128 * 2);
  _Float16* W2  = (_Float16*)alloc((size_t)128 * 512 * 2);
  _Float16* qkv = (_Float16*)alloc((size_t)LTOT * 384 * 2);
  float*    att = (float*)   alloc((size_t)LTOT * CC * 4);
  _Float16* h1  = (_Float16*)alloc((size_t)LTOT * HIDN * 2);

  wconv_kernel<<<256, 256, 0, stream>>>(qkv_w, proj_w, fc1_w, fc2_w, Wq, Wp, W1, W2);
  ln_qkv_kernel<<<LTOT / 16, 256, 0, stream>>>(x, n1w, n1b, Wq, qkv);
  attn_kernel<<<dim3(64, 4), 256, 0, stream>>>(qkv, att, 32, 4, 0);   // branch 0
  attn_kernel<<<dim3(64, 4), 256, 0, stream>>>(qkv, att, 4, 32, 64);  // branch 1
  lepe_kernel<<<64, 256, 0, stream>>>(qkv, cw0, cb0, att, 32, 4, 0);
  lepe_kernel<<<64, 256, 0, stream>>>(qkv, cw1, cb1, att, 4, 32, 64);
  proj_kernel<<<LTOT / 16, 256, 0, stream>>>(att, Wp, proj_b, x, out);
  ln2_fc1_kernel<<<LTOT / 16, 256, 0, stream>>>(out, n2w, n2b, W1, fc1_b, h1);
  fc2_kernel<<<LTOT / 16, 256, 0, stream>>>(h1, W2, fc2_b, out);
}
